// Attention_2370821947593
// MI455X (gfx1250) — matrix operand order repaired and checked by reading, not compiled, not run on hardware
//
#include <hip/hip_runtime.h>
#include <hip/hip_bf16.h>

#define EMB   1024
#define QD    1024
#define CTX   2048
#define BATCH 8
#define MROWS (BATCH * CTX)
#define KC    64   // k-chunk staged in LDS per GEMM stage

typedef _Float16 half16 __attribute__((ext_vector_type(16)));
typedef _Float16 half8  __attribute__((ext_vector_type(8)));
typedef float    f32x8  __attribute__((ext_vector_type(8)));

#define WMMA_F16(a, b, c) \
  __builtin_amdgcn_wmma_f32_16x16x32_f16(false, (a), false, (b), (short)0, (c), false, false)

// ---------------------------------------------------------------------------
// WMMA fragment loaders (layouts per CDNA5 ISA 7.12.2, 16-bit data, wave32)
// A 16x32: lanes 0-15 -> M=lane; V0..3 = K(h*8)..(h*8+7), V4..7 = K(16+h*8)..
// ---------------------------------------------------------------------------
__device__ __forceinline__ half16 load_frag_a(const _Float16* __restrict__ base,
                                              int row0, int k0, int ld) {
  const int lane = threadIdx.x & 31;
  const int r = lane & 15, h = lane >> 4;
  const _Float16* p = base + (size_t)(row0 + r) * ld + (k0 + h * 8);
  half8 lo = *(const half8*)p;         // K = k0 + h*8 .. +7
  half8 hi = *(const half8*)(p + 16);  // K = k0 + 16 + h*8 .. +7
  half16 f;
#pragma unroll
  for (int i = 0; i < 8; ++i) { f[i] = lo[i]; f[i + 8] = hi[i]; }
  return f;
}

// B 32x16: lanes 0-15 -> N=lane; V0..3 = K(h*8)..(h*8+7), V4..7 = K(16+h*8)..
// Source element [n][k], leading dim ld.
__device__ __forceinline__ half16 load_frag_b(const _Float16* __restrict__ base,
                                              int col0, int k0, int ld) {
  const int lane = threadIdx.x & 31;
  const int n = lane & 15, h = lane >> 4;
  const _Float16* p = base + (size_t)(col0 + n) * ld + (k0 + h * 8);
  half8 lo = *(const half8*)p;         // K = k0 + h*8 .. +7
  half8 hi = *(const half8*)(p + 16);  // K = k0 + 16 + h*8 .. +7
  half16 f;
#pragma unroll
  for (int i = 0; i < 8; ++i) { f[i] = lo[i]; f[i + 8] = hi[i]; }
  return f;
}

// LDS variants with fixed leading dim KC
__device__ __forceinline__ half16 lds_frag_a(const _Float16* base, int row0, int k0) {
  const int lane = threadIdx.x & 31;
  const int r = lane & 15, h = lane >> 4;
  const _Float16* p = base + (row0 + r) * KC + k0 + h * 8;
  half8 lo = *(const half8*)p;
  half8 hi = *(const half8*)(p + 16);
  half16 f;
#pragma unroll
  for (int i = 0; i < 8; ++i) { f[i] = lo[i]; f[i + 8] = hi[i]; }
  return f;
}
__device__ __forceinline__ half16 lds_frag_b(const _Float16* base, int col0, int k0) {
  const int lane = threadIdx.x & 31;
  const int n = lane & 15, h = lane >> 4;
  const _Float16* p = base + (col0 + n) * KC + k0 + h * 8;
  half8 lo = *(const half8*)p;
  half8 hi = *(const half8*)(p + 16);
  half16 f;
#pragma unroll
  for (int i = 0; i < 8; ++i) { f[i] = lo[i]; f[i + 8] = hi[i]; }
  return f;
}

// ---------------------------------------------------------------------------
// Async global->LDS copy of a 128x64 fp16 tile (CDNA5 ASYNCcnt path).
// 256 threads x 4 x b128 = 16 KB. LDS dest address = low 32 bits of generic ptr.
// ---------------------------------------------------------------------------
__device__ __forceinline__ void async_copy_tile_128x64(
    const _Float16* __restrict__ g, const _Float16* l, int ld) {
  const int tid = threadIdx.x;
#pragma unroll
  for (int j = 0; j < 4; ++j) {
    const int chunk = tid + 256 * j;       // 1024 chunks of 8 halves
    const int row = chunk >> 3;            // 8 chunks per 64-half row
    const int col = (chunk & 7) * 8;
    const _Float16* gp = g + (size_t)row * ld + col;
    const unsigned lp = (unsigned)(size_t)(l + row * KC + col);
    asm volatile("global_load_async_to_lds_b128 %0, %1, off"
                 :: "v"(lp), "v"(gp)
                 : "memory");
  }
}

__device__ __forceinline__ void wait_async_copies() {
  asm volatile("s_wait_asynccnt 0x0" ::: "memory");
}

// ---------------------------------------------------------------------------
// Stage 0: conversions
// ---------------------------------------------------------------------------
__global__ void convert_x_kernel(const float* __restrict__ x,
                                 _Float16* __restrict__ xh, int n) {
  int idx = blockIdx.x * blockDim.x + threadIdx.x;
  if (idx < n) xh[idx] = (_Float16)x[idx];
}

// Wt[n][e] = W[e][n]  (W row-major [EMB][QD])
__global__ void convert_wt_kernel(const float* __restrict__ W,
                                  _Float16* __restrict__ Wt) {
  int idx = blockIdx.x * blockDim.x + threadIdx.x;  // over EMB*QD
  int e = idx >> 10;
  int n = idx & 1023;
  Wt[(size_t)n * EMB + e] = (_Float16)W[idx];
}

// ---------------------------------------------------------------------------
// Stage 1: projection GEMM  Out[m][n] = sum_e X[m][e] * Wt[n][e] + bias[n]
// Block tile 128(M) x 128(N); 8 waves as 4(M) x 2(N), each wave 32x64 =
// 2x4 WMMA tiles. K staged in LDS (double-buffered async DMA, KC=64).
// All 6 fragments of a k-step are materialized before the 8-WMMA burst so
// the ds_load clause issues once and waits amortize across the burst.
// transposeOut=1 stores V transposed per batch: vt[b][n][c].
// ---------------------------------------------------------------------------
__global__ __launch_bounds__(256) void qkv_gemm_kernel(
    const _Float16* __restrict__ X, const _Float16* __restrict__ Wt,
    const float* __restrict__ bias, _Float16* __restrict__ Out,
    int transposeOut) {
  __shared__ __attribute__((aligned(16))) _Float16 Xs[2][128 * KC];  // 32 KB
  __shared__ __attribute__((aligned(16))) _Float16 Ws[2][128 * KC];  // 32 KB

  const int wave = threadIdx.x >> 5;
  const int wave_m = wave >> 1;   // 0..3
  const int wave_n = wave & 1;    // 0..1
  const int m0 = blockIdx.x * 128;
  const int n0 = blockIdx.y * 128;
  const _Float16* Xblk = X + (size_t)m0 * EMB;
  const _Float16* Wblk = Wt + (size_t)n0 * EMB;

  // prologue: stage k-chunk 0
  async_copy_tile_128x64(Xblk, &Xs[0][0], EMB);
  async_copy_tile_128x64(Wblk, &Ws[0][0], EMB);
  wait_async_copies();
  __syncthreads();

  f32x8 acc[2][4] = {};
  for (int kc = 0; kc < EMB; kc += KC) {
    const int buf = (kc / KC) & 1;
    if (kc + KC < EMB) {  // stage next chunk while computing on current
      async_copy_tile_128x64(Xblk + kc + KC, &Xs[buf ^ 1][0], EMB);
      async_copy_tile_128x64(Wblk + kc + KC, &Ws[buf ^ 1][0], EMB);
    }
    const _Float16* xs = &Xs[buf][0];
    const _Float16* wsl = &Ws[buf][0];
#pragma unroll
    for (int kk = 0; kk < KC; kk += 32) {
      // one ds_load clause: 12 x b128 into distinct registers
      half16 a0  = lds_frag_a(xs, wave_m * 32, kk);
      half16 a1  = lds_frag_a(xs, wave_m * 32 + 16, kk);
      half16 bf0 = lds_frag_b(wsl, wave_n * 64 +  0, kk);
      half16 bf1 = lds_frag_b(wsl, wave_n * 64 + 16, kk);
      half16 bf2 = lds_frag_b(wsl, wave_n * 64 + 32, kk);
      half16 bf3 = lds_frag_b(wsl, wave_n * 64 + 48, kk);
      // 8-WMMA burst
      acc[0][0] = WMMA_F16(a0, bf0, acc[0][0]);
      acc[1][0] = WMMA_F16(a1, bf0, acc[1][0]);
      acc[0][1] = WMMA_F16(a0, bf1, acc[0][1]);
      acc[1][1] = WMMA_F16(a1, bf1, acc[1][1]);
      acc[0][2] = WMMA_F16(a0, bf2, acc[0][2]);
      acc[1][2] = WMMA_F16(a1, bf2, acc[1][2]);
      acc[0][3] = WMMA_F16(a0, bf3, acc[0][3]);
      acc[1][3] = WMMA_F16(a1, bf3, acc[1][3]);
    }
    wait_async_copies();   // next buffer landed
    __syncthreads();       // everyone done reading current buffer
  }

  const int lane = threadIdx.x & 31;
  const int ln = lane & 15, lh = lane >> 4;
#pragma unroll
  for (int t = 0; t < 2; ++t) {
#pragma unroll
    for (int j = 0; j < 4; ++j) {
      const int n = n0 + wave_n * 64 + j * 16 + ln;
      const float bv = bias[n];
#pragma unroll
      for (int r = 0; r < 8; ++r) {
        const int m = m0 + wave_m * 32 + t * 16 + r + 8 * lh;
        const float v = acc[t][j][r] + bv;
        size_t idx;
        if (!transposeOut) {
          idx = (size_t)m * QD + n;
        } else {
          const int bb = m / CTX, c = m % CTX;
          idx = ((size_t)bb * QD + n) * CTX + c;
        }
        Out[idx] = (_Float16)v;
      }
    }
  }
}

// ---------------------------------------------------------------------------
// Stage 2: fused causal attention.  One block = (batch b, 16-query strip qt).
// Phase 1: score strip S[16][2048] via WMMA -> LDS (fp16); each wave owns
//          k-tiles kt = wave + 8*i, processed in groups of 4 so one Q
//          A-fragment feeds 4 WMMAs (fragments preloaded before the burst).
// Phase 2: row softmax in LDS.
// Phase 3: O[16][1024] = P . V (V pre-transposed), 8 waves x 128 cols.
// ---------------------------------------------------------------------------
__global__ __launch_bounds__(256) void attn_kernel(
    const _Float16* __restrict__ qh, const _Float16* __restrict__ kh,
    const _Float16* __restrict__ vt, float* __restrict__ out) {
  __shared__ __attribute__((aligned(16))) _Float16 P[16][CTX];  // 64 KB

  const int bid = blockIdx.x;
  const int b = bid >> 7;          // / (CTX/16)
  const int qt = bid & 127;
  const int q0 = qt * 16;
  const _Float16* qb = qh + (size_t)b * CTX * QD;
  const _Float16* kb = kh + (size_t)b * CTX * QD;
  const _Float16* vb = vt + (size_t)b * QD * CTX;

  const int wave = threadIdx.x >> 5;
  const int lane = threadIdx.x & 31;
  const int ln = lane & 15, lh = lane >> 4;
  const float scale = 0.03125f;  // 1/sqrt(1024)

  // ---- Phase 1: scores (groups of 4 k-tiles per wave) ----
#pragma unroll
  for (int g = 0; g < 4; ++g) {
    const int kt_base = wave + 32 * g;  // tiles kt_base + 8*t, t=0..3
    if (kt_base > qt) {                 // whole group above diagonal
#pragma unroll
      for (int t = 0; t < 4; ++t) {
        const int kt = kt_base + 8 * t;
#pragma unroll
        for (int r = 0; r < 8; ++r)
          P[r + 8 * lh][kt * 16 + ln] = (_Float16)(-30000.0f);
      }
      continue;
    }
    f32x8 acc[4] = {};
    for (int k0 = 0; k0 < QD; k0 += 32) {
      // preload all 5 fragments, then the 4-WMMA burst
      half16 a   = load_frag_a(qb, q0, k0, QD);
      half16 bf0 = load_frag_b(kb, (kt_base +  0) * 16, k0, QD);
      half16 bf1 = load_frag_b(kb, (kt_base +  8) * 16, k0, QD);
      half16 bf2 = load_frag_b(kb, (kt_base + 16) * 16, k0, QD);
      half16 bf3 = load_frag_b(kb, (kt_base + 24) * 16, k0, QD);
      acc[0] = WMMA_F16(a, bf0, acc[0]);
      acc[1] = WMMA_F16(a, bf1, acc[1]);
      acc[2] = WMMA_F16(a, bf2, acc[2]);
      acc[3] = WMMA_F16(a, bf3, acc[3]);
    }
#pragma unroll
    for (int t = 0; t < 4; ++t) {
      const int kt = kt_base + 8 * t;
#pragma unroll
      for (int r = 0; r < 8; ++r) {
        const int m = r + 8 * lh;
        const int key = kt * 16 + ln;
        float s = acc[t][r] * scale;
        if (key > q0 + m) s = -30000.0f;  // causal mask (covers full tiles too)
        P[m][kt * 16 + ln] = (_Float16)s;
      }
    }
  }
  __syncthreads();

  // ---- Phase 2: softmax (16 threads per row, width-16 shuffles) ----
  {
    const int row = threadIdx.x >> 4;
    const int c0 = threadIdx.x & 15;
    float mx = -1e30f;
    for (int c = c0; c < CTX; c += 16) mx = fmaxf(mx, (float)P[row][c]);
#pragma unroll
    for (int off = 8; off > 0; off >>= 1)
      mx = fmaxf(mx, __shfl_xor(mx, off, 16));
    float sum = 0.f;
    for (int c = c0; c < CTX; c += 16) {
      float e = __expf((float)P[row][c] - mx);
      P[row][c] = (_Float16)e;
      sum += e;
    }
#pragma unroll
    for (int off = 8; off > 0; off >>= 1) sum += __shfl_xor(sum, off, 16);
    const float inv = 1.0f / sum;
    for (int c = c0; c < CTX; c += 16)
      P[row][c] = (_Float16)((float)P[row][c] * inv);
  }
  __syncthreads();

  // ---- Phase 3: O = P . V ----
  const int n0 = wave * 128;
  f32x8 acc[8] = {};
  const int kmax = ((qt + 1) * 16 + 31) & ~31;  // causal skip (rounded to 32)
  for (int kk = 0; kk < kmax; kk += 32) {
    half16 a;
    {  // A-frag from LDS (ds_load_b128)
      const _Float16* p0 = &P[ln][kk + lh * 8];
      half8 lo = *(const half8*)p0;
      half8 hi = *(const half8*)(p0 + 16);
#pragma unroll
      for (int i = 0; i < 8; ++i) { a[i] = lo[i]; a[i + 8] = hi[i]; }
    }
    // preload V fragments in two groups of 4 to bound register pressure
    half16 bf0 = load_frag_b(vb, n0 +  0, kk, CTX);
    half16 bf1 = load_frag_b(vb, n0 + 16, kk, CTX);
    half16 bf2 = load_frag_b(vb, n0 + 32, kk, CTX);
    half16 bf3 = load_frag_b(vb, n0 + 48, kk, CTX);
    acc[0] = WMMA_F16(a, bf0, acc[0]);
    acc[1] = WMMA_F16(a, bf1, acc[1]);
    acc[2] = WMMA_F16(a, bf2, acc[2]);
    acc[3] = WMMA_F16(a, bf3, acc[3]);
    half16 bf4 = load_frag_b(vb, n0 +  64, kk, CTX);
    half16 bf5 = load_frag_b(vb, n0 +  80, kk, CTX);
    half16 bf6 = load_frag_b(vb, n0 +  96, kk, CTX);
    half16 bf7 = load_frag_b(vb, n0 + 112, kk, CTX);
    acc[4] = WMMA_F16(a, bf4, acc[4]);
    acc[5] = WMMA_F16(a, bf5, acc[5]);
    acc[6] = WMMA_F16(a, bf6, acc[6]);
    acc[7] = WMMA_F16(a, bf7, acc[7]);
  }
#pragma unroll
  for (int j = 0; j < 8; ++j) {
#pragma unroll
    for (int r = 0; r < 8; ++r) {
      const int m = q0 + r + 8 * lh;
      const int n = n0 + j * 16 + ln;
      out[((size_t)b * CTX + m) * QD + n] = acc[j][r];
    }
  }
}

// ---------------------------------------------------------------------------
extern "C" void kernel_launch(void* const* d_in, const int* in_sizes, int n_in,
                              void* d_out, int out_size, void* d_ws,
                              size_t ws_size, hipStream_t stream) {
  (void)in_sizes; (void)n_in; (void)out_size; (void)ws_size;
  const float* x  = (const float*)d_in[0];
  const float* Qw = (const float*)d_in[1];
  const float* Qb = (const float*)d_in[2];
  const float* Kw = (const float*)d_in[3];
  const float* Kb = (const float*)d_in[4];
  const float* Vw = (const float*)d_in[5];
  const float* Vb = (const float*)d_in[6];

  // Workspace layout (fp16 elements), total 67M halves = 134 MB
  _Float16* ws = (_Float16*)d_ws;
  const size_t MEGA = 1024 * 1024;
  _Float16* xh  = ws;              // 16M  x  [M][EMB]
  _Float16* wtq = ws + 16 * MEGA;  //  1M  Qw^T [QD][EMB]
  _Float16* wtk = ws + 17 * MEGA;
  _Float16* wtv = ws + 18 * MEGA;
  _Float16* qh  = ws + 19 * MEGA;  // 16M  q [b][c][d]
  _Float16* kh  = ws + 35 * MEGA;  // 16M  k [b][c][d]
  _Float16* vth = ws + 51 * MEGA;  // 16M  v^T [b][d][c]

  const int nx = MROWS * EMB;  // 16,777,216
  convert_x_kernel<<<nx / 256, 256, 0, stream>>>(x, xh, nx);
  convert_wt_kernel<<<(EMB * QD) / 256, 256, 0, stream>>>(Qw, wtq);
  convert_wt_kernel<<<(EMB * QD) / 256, 256, 0, stream>>>(Kw, wtk);
  convert_wt_kernel<<<(EMB * QD) / 256, 256, 0, stream>>>(Vw, wtv);

  dim3 g(MROWS / 128, QD / 128);  // (128, 8)
  qkv_gemm_kernel<<<g, 256, 0, stream>>>(xh, wtq, Qb, qh, 0);
  qkv_gemm_kernel<<<g, 256, 0, stream>>>(xh, wtk, Kb, kh, 0);
  qkv_gemm_kernel<<<g, 256, 0, stream>>>(xh, wtv, Vb, vth, 1);

  attn_kernel<<<BATCH * (CTX / 16), 256, 0, stream>>>(
      qh, kh, vth, (float*)d_out);
}
